// INTP_Model_13357348290612
// MI455X (gfx1250) — matrix operand order repaired and checked by reading, not compiled, not run on hardware
//
#include <hip/hip_runtime.h>
#include <hip/hip_bf16.h>

// B=4, L=512, F_OP=10, F_EU=2, D=128, H=128
#define Bb 4
#define Ll 512
#define Dd 128
#define NROW (Bb*Ll)          // 2048
#define DEG_INV (1.0f/511.0f)
#define ADJ_SCALE 0.08838834764831845f  // 1/sqrt(128)

typedef __attribute__((ext_vector_type(16))) __bf16 v16bf;
typedef __attribute__((ext_vector_type(8)))  __bf16 v8bf;
typedef __attribute__((ext_vector_type(8)))  float  v8f;

// ---- WMMA fragment loaders, per CDNA5 ISA 7.12.2 layouts (wave32) ----
// A 16x32 bf16 (row-major source, leading dim ld):
//   lane m(<16): K[0..7] then K[16..23]; lane 16+m: K[8..15] then K[24..31]
static __device__ inline v16bf load_matA(const __bf16* base, int row0, int ld, int k0) {
  int lid = threadIdx.x & 31;
  int m = lid & 15, half = lid >> 4;
  const __bf16* p = base + (size_t)(row0 + m) * ld + k0;
  v8bf lo = *(const v8bf*)(p + half * 8);
  v8bf hi = *(const v8bf*)(p + 16 + half * 8);
  v16bf r;
#pragma unroll
  for (int i = 0; i < 8; ++i) { r[i] = lo[i]; r[i + 8] = hi[i]; }
  return r;
}
// B 32x16 bf16 from an N-major (N x K row-major) source: column n = source row n.
//   lane n(<16): K[0..7] then K[16..23]; lane 16+n: K[8..15] then K[24..31]
static __device__ inline v16bf load_matB(const __bf16* base, int n0, int ld, int k0) {
  int lid = threadIdx.x & 31;
  int n = lid & 15, half = lid >> 4;
  const __bf16* p = base + (size_t)(n0 + n) * ld + k0;
  v8bf lo = *(const v8bf*)(p + half * 8);
  v8bf hi = *(const v8bf*)(p + 16 + half * 8);
  v16bf r;
#pragma unroll
  for (int i = 0; i < 8; ++i) { r[i] = lo[i]; r[i + 8] = hi[i]; }
  return r;
}

static __device__ inline v8f wmma_bf16(v16bf a, v16bf b, v8f c) {
  return __builtin_amdgcn_wmma_f32_16x16x32_bf16(false, a, false, b, (short)0, c,
                                                 false, false);
}

static __device__ inline unsigned long long shflxor_u64(unsigned long long v, int m) {
  unsigned int lo = (unsigned int)v, hi = (unsigned int)(v >> 32);
  lo = __shfl_xor(lo, m, 32);
  hi = __shfl_xor(hi, m, 32);
  return ((unsigned long long)hi << 32) | lo;
}

// ---- K0: transpose weights to bf16 (N-major) + init per-row argmin slots ----
__global__ void k_init(const float* Wa, const float* Wb, const float* W1l,
                       const float* W1r, __bf16* Wt, unsigned long long* rowMin) {
  int idx = blockIdx.x * 256 + threadIdx.x;   // 0..65535
  const float* srcs[4] = {Wa, Wb, W1l, W1r};
  int w = idx >> 14;
  int rem = idx & 16383;
  int k = rem >> 7, n = rem & 127;
  Wt[w * 16384 + n * 128 + k] = (__bf16)srcs[w][k * 128 + n];
  if (idx < NROW) rowMin[idx] = ~0ull;
}

// ---- K1: embeddings (K=12 dot), x = i_emb - ic_emb ----
__global__ void k_emb(const float* __restrict__ inputs, const float* __restrict__ coords,
                      const float* __restrict__ ccv, const float* Wc, const float* bcv,
                      const float* Wcc, const float* bccv, __bf16* i_emb_bf,
                      __bf16* ic_emb_bf, float* x_f, __bf16* x_bf) {
  int r = blockIdx.x;        // 0..2047
  int d = threadIdx.x;       // 0..127
  int b = r >> 9;
  const float* inr = inputs + r * 10;
  const float* in0 = inputs + (b << 9) * 10;
  float ie = bcv[d], ice = bccv[d];
#pragma unroll
  for (int k = 0; k < 10; ++k) {
    ie  += inr[k] * Wc[k * 128 + d];
    ice += (inr[k] - in0[k]) * Wcc[k * 128 + d];
  }
#pragma unroll
  for (int k = 0; k < 2; ++k) {
    ie  += coords[r * 2 + k] * Wc[(10 + k) * 128 + d];
    ice += ccv[r * 2 + k]    * Wcc[(10 + k) * 128 + d];
  }
  float xv = ie - ice;
  i_emb_bf[r * 128 + d]  = (__bf16)ie;
  ic_emb_bf[r * 128 + d] = (__bf16)ice;
  x_f[r * 128 + d] = xv;
  x_bf[r * 128 + d] = (__bf16)xv;
}

// ---- K2: four WMMA projections a / bt / a_c / b_c = emb @ {Wa,Wb} + bias ----
__global__ void k_proj(const __bf16* i_emb, const __bf16* ic_emb, const __bf16* Wt,
                       const float* ba, const float* bb, __bf16* a_bf, __bf16* bt_bf,
                       __bf16* ac_bf, __bf16* bc_bf) {
  int wave = threadIdx.x >> 5;
  int t = blockIdx.x * 8 + wave;    // 0..4095
  int o = t >> 10;                  // which of the 4 outputs
  int rem = t & 1023;
  int ri = rem >> 3, ci = rem & 7;
  const __bf16* A  = (o < 2) ? i_emb : ic_emb;
  const __bf16* Bw = Wt + ((o & 1) ? 16384 : 0);
  const float* bias = (o & 1) ? bb : ba;
  __bf16* dst = (o == 0) ? a_bf : (o == 1) ? bt_bf : (o == 2) ? ac_bf : bc_bf;
  int r0 = ri * 16, c0 = ci * 16;
  v8f acc = {0, 0, 0, 0, 0, 0, 0, 0};
#pragma unroll
  for (int kk = 0; kk < 4; ++kk)
    acc = wmma_bf16(load_matA(A, r0, 128, kk * 32), load_matB(Bw, c0, 128, kk * 32), acc);
  int lid = threadIdx.x & 31;
  int col = c0 + (lid & 15);
  int half = lid >> 4;
  float bv = bias[col];
#pragma unroll
  for (int v = 0; v < 8; ++v)
    dst[(r0 + v + 8 * half) * 128 + col] = (__bf16)(acc[v] + bv);
}

// ---- K3: adj & adj_c 16x16 tiles (WMMA), diff partials, per-row argmin ----
__global__ void k_adj(const __bf16* a_bf, const __bf16* bt_bf, const __bf16* ac_bf,
                      const __bf16* bc_bf, unsigned long long* rowMin,
                      float* diffPartial) {
  __shared__ float sred[256];
  int wave = threadIdx.x >> 5;
  int lid = threadIdx.x & 31;
  int t = blockIdx.x * 8 + wave;    // 0..4095 (4 batches x 32x32 tiles)
  int b = t >> 10;
  int rem = t & 1023;
  int ti = rem >> 5, tj = rem & 31;
  int gr0 = b * 512 + ti * 16;      // adj rows (global)
  int gc0 = b * 512 + tj * 16;      // adj cols -> rows of bt
  v8f acc  = {0, 0, 0, 0, 0, 0, 0, 0};
  v8f accc = {0, 0, 0, 0, 0, 0, 0, 0};
#pragma unroll
  for (int kk = 0; kk < 4; ++kk) {
    acc  = wmma_bf16(load_matA(a_bf,  gr0, 128, kk * 32),
                     load_matB(bt_bf, gc0, 128, kk * 32), acc);
    accc = wmma_bf16(load_matA(ac_bf, gr0, 128, kk * 32),
                     load_matB(bc_bf, gc0, 128, kk * 32), accc);
  }
  int half = lid >> 4;
  int colLocal = tj * 16 + (lid & 15);
  float local = 0.f;
#pragma unroll
  for (int v = 0; v < 8; ++v) {
    float Av  = acc[v]  * ADJ_SCALE;
    float Acv = accc[v] * ADJ_SCALE;
    float e = Av - Acv;
    local += e * e;
    // order-preserving u32 key for float min; low word ~col picks the
    // largest index among equal minima (jax top_k tie rule for the excluded one)
    unsigned int u = __float_as_uint(Av);
    unsigned int key = (u & 0x80000000u) ? ~u : (u | 0x80000000u);
    unsigned long long packed =
        ((unsigned long long)key << 32) | (unsigned int)(~(unsigned int)colLocal);
#pragma unroll
    for (int off = 1; off < 16; off <<= 1) {
      unsigned long long other = shflxor_u64(packed, off);
      if (other < packed) packed = other;
    }
    if ((lid & 15) == 0)
      atomicMin(&rowMin[gr0 + v + 8 * half], packed);   // deterministic (min)
  }
  sred[threadIdx.x] = local;
  __syncthreads();
  for (int s = 128; s > 0; s >>= 1) {
    if (threadIdx.x < (unsigned)s) sred[threadIdx.x] += sred[threadIdx.x + s];
    __syncthreads();
  }
  if (threadIdx.x == 0) diffPartial[blockIdx.x] = sred[0];
}

// ---- K4: per-batch column sums of x ----
__global__ void k_sumx(const float* x_f, float* S) {
  int b = blockIdx.x, d = threadIdx.x;
  float s = 0.f;
  const float* p = x_f + (size_t)(b << 9) * 128 + d;
  for (int l = 0; l < 512; ++l) s += p[l * 128];
  S[b * 128 + d] = s;
}

// ---- K4b: agg1[r] = (S[b] - x[argmin_r]) / 511 ----
__global__ void k_agg1(const float* x_f, const float* S,
                       const unsigned long long* rowMin, __bf16* agg1_bf) {
  int r = blockIdx.x, d = threadIdx.x;
  int b = r >> 9;
  unsigned int colLocal = ~(unsigned int)(rowMin[r] & 0xFFFFFFFFull);
  int m = (b << 9) + (int)colLocal;
  float v = (S[b * 128 + d] - x_f[(size_t)m * 128 + d]) * DEG_INV;
  agg1_bf[r * 128 + d] = (__bf16)v;
}

// ---- K5: h1 = relu(agg1 @ W1l + b1l + x @ W1r)  (two fused WMMA GEMMs) ----
__global__ void k_conv1(const __bf16* agg1_bf, const __bf16* x_bf, const __bf16* Wt,
                        const float* b1l, float* h1) {
  int wave = threadIdx.x >> 5;
  int t = blockIdx.x * 8 + wave;    // 0..1023
  int ri = t >> 3, ci = t & 7;
  int r0 = ri * 16, c0 = ci * 16;
  const __bf16* W1l_t = Wt + 2 * 16384;
  const __bf16* W1r_t = Wt + 3 * 16384;
  v8f acc = {0, 0, 0, 0, 0, 0, 0, 0};
#pragma unroll
  for (int kk = 0; kk < 4; ++kk)
    acc = wmma_bf16(load_matA(agg1_bf, r0, 128, kk * 32),
                    load_matB(W1l_t, c0, 128, kk * 32), acc);
#pragma unroll
  for (int kk = 0; kk < 4; ++kk)
    acc = wmma_bf16(load_matA(x_bf, r0, 128, kk * 32),
                    load_matB(W1r_t, c0, 128, kk * 32), acc);
  int lid = threadIdx.x & 31;
  int half = lid >> 4;
  int col = c0 + (lid & 15);
  float bv = b1l[col];
#pragma unroll
  for (int v = 0; v < 8; ++v) {
    float h = acc[v] + bv;
    h1[(size_t)(r0 + v + 8 * half) * 128 + col] = h > 0.f ? h : 0.f;
  }
}

// ---- K6: blocks 0..3 -> out_head + target_head; block 4 -> diff reduce ----
__global__ void k_final(const float* h1, const unsigned long long* rowMin,
                        const float* W2l, const float* b2l, const float* W2r,
                        const float* targets, const float* diffPartial, float* out) {
  __shared__ float sred[128];
  int d = threadIdx.x;
  if (blockIdx.x < 4) {
    int b = blockIdx.x;
    float s = 0.f;
    const float* p = h1 + (size_t)(b << 9) * 128 + d;
    for (int l = 0; l < 512; ++l) s += p[l * 128];
    unsigned int colLocal = ~(unsigned int)(rowMin[b << 9] & 0xFFFFFFFFull);
    int m = (b << 9) + (int)colLocal;
    float agg2 = (s - h1[(size_t)m * 128 + d]) * DEG_INV;
    sred[d] = agg2 * W2l[d] + h1[(size_t)(b << 9) * 128 + d] * W2r[d];
    __syncthreads();
    for (int st = 64; st > 0; st >>= 1) {
      if (d < st) sred[d] += sred[d + st];
      __syncthreads();
    }
    if (d == 0) {
      out[b] = sred[0] + b2l[0];       // out_head
      out[5 + b] = targets[b << 9];    // target_head
    }
  } else {
    float s = diffPartial[d] + diffPartial[d + 128] + diffPartial[d + 256] +
              diffPartial[d + 384];
    sred[d] = s;
    __syncthreads();
    for (int st = 64; st > 0; st >>= 1) {
      if (d < st) sred[d] += sred[d + st];
      __syncthreads();
    }
    if (d == 0) out[4] = sred[0] * (1.0f / 1048576.0f);  // mean over 4*512*512
  }
}

extern "C" void kernel_launch(void* const* d_in, const int* in_sizes, int n_in,
                              void* d_out, int out_size, void* d_ws, size_t ws_size,
                              hipStream_t stream) {
  const float* inputs  = (const float*)d_in[0];
  const float* coords  = (const float*)d_in[1];
  const float* targets = (const float*)d_in[2];
  const float* ccv     = (const float*)d_in[3];
  // d_in[4] = input_lenths (unused by reference _forward)
  const float* Wc  = (const float*)d_in[5];
  const float* bc  = (const float*)d_in[6];
  const float* Wcc = (const float*)d_in[7];
  const float* bcc = (const float*)d_in[8];
  const float* Wa  = (const float*)d_in[9];
  const float* ba  = (const float*)d_in[10];
  const float* Wb  = (const float*)d_in[11];
  const float* bb  = (const float*)d_in[12];
  const float* W1l = (const float*)d_in[13];
  const float* b1l = (const float*)d_in[14];
  const float* W1r = (const float*)d_in[15];
  const float* W2l = (const float*)d_in[16];
  const float* b2l = (const float*)d_in[17];
  const float* W2r = (const float*)d_in[18];

  char* w = (char*)d_ws;
  auto carve = [&](size_t n) -> char* {
    char* p = w;
    w += (n + 255) & ~size_t(255);
    return p;
  };
  __bf16* i_emb_bf  = (__bf16*)carve((size_t)NROW * 128 * 2);
  __bf16* ic_emb_bf = (__bf16*)carve((size_t)NROW * 128 * 2);
  float*  x_f       = (float*) carve((size_t)NROW * 128 * 4);
  __bf16* x_bf      = (__bf16*)carve((size_t)NROW * 128 * 2);
  __bf16* a_bf      = (__bf16*)carve((size_t)NROW * 128 * 2);
  __bf16* bt_bf     = (__bf16*)carve((size_t)NROW * 128 * 2);
  __bf16* ac_bf     = (__bf16*)carve((size_t)NROW * 128 * 2);
  __bf16* bc_bf     = (__bf16*)carve((size_t)NROW * 128 * 2);
  __bf16* Wt        = (__bf16*)carve((size_t)4 * 128 * 128 * 2);
  unsigned long long* rowMin = (unsigned long long*)carve((size_t)NROW * 8);
  float*  S         = (float*) carve((size_t)Bb * 128 * 4);
  __bf16* agg1_bf   = (__bf16*)carve((size_t)NROW * 128 * 2);
  float*  h1        = (float*) carve((size_t)NROW * 128 * 4);
  float*  diffPartial = (float*)carve((size_t)512 * 4);

  k_init<<<256, 256, 0, stream>>>(Wa, Wb, W1l, W1r, Wt, rowMin);
  k_emb<<<NROW, 128, 0, stream>>>(inputs, coords, ccv, Wc, bc, Wcc, bcc, i_emb_bf,
                                  ic_emb_bf, x_f, x_bf);
  k_proj<<<512, 256, 0, stream>>>(i_emb_bf, ic_emb_bf, Wt, ba, bb, a_bf, bt_bf,
                                  ac_bf, bc_bf);
  k_adj<<<512, 256, 0, stream>>>(a_bf, bt_bf, ac_bf, bc_bf, rowMin, diffPartial);
  k_sumx<<<Bb, 128, 0, stream>>>(x_f, S);
  k_agg1<<<NROW, 128, 0, stream>>>(x_f, S, rowMin, agg1_bf);
  k_conv1<<<128, 256, 0, stream>>>(agg1_bf, x_bf, Wt, b1l, h1);
  k_final<<<5, 128, 0, stream>>>(h1, rowMin, W2l, b2l, W2r, targets, diffPartial,
                                 (float*)d_out);
}
